// ExtractorLSTM_19490561589341
// MI455X (gfx1250) — matrix operand order repaired and checked by reading, not compiled, not run on hardware
//
#include <hip/hip_runtime.h>
#include <hip/hip_bf16.h>
#include <math.h>

// ---------------------------------------------------------------------------
// Problem sizes (fixed by the reference)
// ---------------------------------------------------------------------------
#define BB   16
#define SS   512
#define BS   8192          // B*S = sequential scan length
#define II   768
#define HH   1536
#define G4   6144          // 4*H
#define RWG  96            // persistent workgroups in recurrence (96*16 = 1536 units)

typedef __attribute__((ext_vector_type(16))) __bf16 v16bf;
typedef __attribute__((ext_vector_type(8)))  float  v8f;

// Vector type matching the async-LDS builtin's parameter (gcc vector_size style)
typedef int v4i_vs __attribute__((vector_size(16)));
typedef __attribute__((address_space(1))) v4i_vs* gptr_v4i;
typedef __attribute__((address_space(3))) v4i_vs* lptr_v4i;

struct alignas(16) U128 { unsigned int x, y, z, w; };

union FragAB { v16bf v; U128 q[2]; };
union FragC  { v8f   v; float f[8]; };

// gfx1250 async global->LDS path (guarded: falls back to ds_store staging)
#if defined(__gfx1250__)
#if __has_builtin(__builtin_amdgcn_global_load_async_to_lds_b128) && \
    __has_builtin(__builtin_amdgcn_s_wait_asynccnt)
#define USE_ASYNC_LDS 1
#endif
#endif

__device__ __forceinline__ unsigned short f32_to_bf16_rne(float f) {
  unsigned int u = __float_as_uint(f);
  u += 0x7FFFu + ((u >> 16) & 1u);            // round to nearest even
  return (unsigned short)(u >> 16);
}
__device__ __forceinline__ float bf16lo_to_f32(unsigned int packed) {
  return __uint_as_float(packed << 16);
}
__device__ __forceinline__ float bf16hi_to_f32(unsigned int packed) {
  return __uint_as_float(packed & 0xFFFF0000u);
}

// ---------------------------------------------------------------------------
// Prologue: f32 -> bf16 conversions
// ---------------------------------------------------------------------------
__global__ __launch_bounds__(256) void cvt_bf16(const float* __restrict__ in,
                                                unsigned short* __restrict__ out,
                                                int n) {
  int i = (blockIdx.x * 256 + threadIdx.x) * 4;
  if (i + 3 < n) {
    out[i + 0] = f32_to_bf16_rne(in[i + 0]);
    out[i + 1] = f32_to_bf16_rne(in[i + 1]);
    out[i + 2] = f32_to_bf16_rne(in[i + 2]);
    out[i + 3] = f32_to_bf16_rne(in[i + 3]);
  }
}

// Swizzle w_hh (f32 [6144][1536]) into the exact streaming order used by the
// recurrence kernel: slot = ((it*RWG + wg)*256 + tid), 8 bf16 per slot.
__global__ __launch_bounds__(256) void swizzle_whh(const float* __restrict__ whh,
                                                   unsigned short* __restrict__ out) {
  int slot = blockIdx.x * 256 + threadIdx.x;     // 0 .. 48*96*256-1
  int tid  = slot & 255;
  int wg   = (slot >> 8) % RWG;
  int it   = slot / (256 * RWG);                 // 0..47
  int wave = tid >> 5, lane = tid & 31;
  int t4   = lane & 3;
  int R    = wave * 8 + (lane >> 2);
  int gate = R & 3, unitL = R >> 2;
  int row  = gate * HH + wg * 16 + unitL;
  int k    = t4 * 384 + it * 8;
  unsigned short* dst = out + (size_t)slot * 8;
#pragma unroll
  for (int j = 0; j < 8; ++j)
    dst[j] = f32_to_bf16_rne(whh[(size_t)row * HH + k + j]);
}

__global__ __launch_bounds__(256) void init_state(float* __restrict__ hbuf,
                                                  int* __restrict__ bar) {
  int i = blockIdx.x * 256 + threadIdx.x;
  if (i < 2 * HH) hbuf[i] = 0.0f;
  if (i < 8)      bar[i]  = 0;
}

// ---------------------------------------------------------------------------
// Phase 1: gx[8192][6144] = x @ w_ih^T + (b_ih + b_hh), bf16 WMMA, f32 accum
// Block = 256 threads (8 waves), tile 128x128, K-step 32, double-buffered
// async global->LDS staging when available.
// ---------------------------------------------------------------------------
#define LDT 40   // padded LDS row stride in bf16 elems (80B: 16B aligned)

__global__ __launch_bounds__(256) void gemm_gx(const unsigned short* __restrict__ xb,
                                               const unsigned short* __restrict__ wihb,
                                               const float* __restrict__ b_ih,
                                               const float* __restrict__ b_hh,
                                               float* __restrict__ gx) {
  __shared__ unsigned short As[2][128 * LDT];
  __shared__ unsigned short Bs[2][128 * LDT];

  const int tid  = threadIdx.x;
  const int wave = tid >> 5, lane = tid & 31;
  const int wm   = wave >> 2;          // 0..1 : 64-row slab
  const int wn   = wave & 3;           // 0..3 : 32-col slab
  const int hi   = lane >> 4;          // half-wave select
  const int l16  = lane & 15;
  const int m0   = blockIdx.y * 128;
  const int n0   = blockIdx.x * 128;

  // Each thread stages two 16B chunks per tile per matrix.
  const int chunk0 = tid * 2, chunk1 = tid * 2 + 1;   // 0..511
  const int r0 = chunk0 >> 2, kc0 = (chunk0 & 3) * 8;
  const int r1 = chunk1 >> 2, kc1 = (chunk1 & 3) * 8;

  FragC acc[4][2];
#pragma unroll
  for (int mi = 0; mi < 4; ++mi)
#pragma unroll
    for (int ni = 0; ni < 2; ++ni)
#pragma unroll
      for (int r = 0; r < 8; ++r) acc[mi][ni].f[r] = 0.0f;

#ifdef USE_ASYNC_LDS
  // -------- async global->LDS double-buffered pipeline --------
  auto issue_tile = [&](int k0, int buf) {
    __builtin_amdgcn_global_load_async_to_lds_b128(
        (gptr_v4i)(xb + (size_t)(m0 + r0) * II + k0 + kc0),
        (lptr_v4i)(&As[buf][r0 * LDT + kc0]), 0, 0);
    __builtin_amdgcn_global_load_async_to_lds_b128(
        (gptr_v4i)(xb + (size_t)(m0 + r1) * II + k0 + kc1),
        (lptr_v4i)(&As[buf][r1 * LDT + kc1]), 0, 0);
    __builtin_amdgcn_global_load_async_to_lds_b128(
        (gptr_v4i)(wihb + (size_t)(n0 + r0) * II + k0 + kc0),
        (lptr_v4i)(&Bs[buf][r0 * LDT + kc0]), 0, 0);
    __builtin_amdgcn_global_load_async_to_lds_b128(
        (gptr_v4i)(wihb + (size_t)(n0 + r1) * II + k0 + kc1),
        (lptr_v4i)(&Bs[buf][r1 * LDT + kc1]), 0, 0);
  };

  issue_tile(0, 0);
  int buf = 0;
  for (int k0 = 0; k0 < II; k0 += 32, buf ^= 1) {
    if (k0 + 32 < II) {
      issue_tile(k0 + 32, buf ^ 1);                // prefetch next tile
      __builtin_amdgcn_s_wait_asynccnt(4);         // oldest 4 (this tile) done
    } else {
      __builtin_amdgcn_s_wait_asynccnt(0);
    }
    __syncthreads();
#else
  int buf = 0;
  for (int k0 = 0; k0 < II; k0 += 32) {
    // -------- fallback: global->VGPR->LDS staging --------
    *(U128*)(&As[0][r0 * LDT + kc0]) =
        *(const U128*)(xb + (size_t)(m0 + r0) * II + k0 + kc0);
    *(U128*)(&As[0][r1 * LDT + kc1]) =
        *(const U128*)(xb + (size_t)(m0 + r1) * II + k0 + kc1);
    *(U128*)(&Bs[0][r0 * LDT + kc0]) =
        *(const U128*)(wihb + (size_t)(n0 + r0) * II + k0 + kc0);
    *(U128*)(&Bs[0][r1 * LDT + kc1]) =
        *(const U128*)(wihb + (size_t)(n0 + r1) * II + k0 + kc1);
    __syncthreads();
#endif

    // Build fragments per the CDNA5 16-bit A (16x32) / B (32x16) VGPR layouts.
    FragAB a[4], b[2];
#pragma unroll
    for (int mi = 0; mi < 4; ++mi) {
      int m = wm * 64 + mi * 16 + l16;
      a[mi].q[0] = *(const U128*)(&As[buf][m * LDT + hi * 8]);       // K = 8hi..
      a[mi].q[1] = *(const U128*)(&As[buf][m * LDT + 16 + hi * 8]);  // K = 16+8hi..
    }
#pragma unroll
    for (int ni = 0; ni < 2; ++ni) {
      int n = wn * 32 + ni * 16 + l16;
      b[ni].q[0] = *(const U128*)(&Bs[buf][n * LDT + hi * 8]);       // K = 8hi..
      b[ni].q[1] = *(const U128*)(&Bs[buf][n * LDT + 16 + hi * 8]);  // K = 16+8hi..
    }

#pragma unroll
    for (int mi = 0; mi < 4; ++mi)
#pragma unroll
      for (int ni = 0; ni < 2; ++ni)
        acc[mi][ni].v = __builtin_amdgcn_wmma_f32_16x16x32_bf16(
            false, a[mi].v, false, b[ni].v, (short)0, acc[mi][ni].v, false, false);
    __syncthreads();
  }

  // Epilogue: fold biases, store f32 gx.  C/D layout: elem r -> M = r + 8*hi.
#pragma unroll
  for (int mi = 0; mi < 4; ++mi)
#pragma unroll
    for (int ni = 0; ni < 2; ++ni) {
      int n = n0 + wn * 32 + ni * 16 + l16;
      float bias = b_ih[n] + b_hh[n];
#pragma unroll
      for (int r = 0; r < 8; ++r) {
        int m = m0 + wm * 64 + mi * 16 + r + 8 * hi;
        gx[(size_t)m * G4 + n] = acc[mi][ni].f[r] + bias;
      }
    }
}

// ---------------------------------------------------------------------------
// Phase 2: 8192 sequential LSTM steps.  Persistent grid of RWG blocks.
// Each block owns 16 hidden units (64 gate rows), 4 lanes per row dot-product.
// bf16 weights streamed from L2 in perfectly coalesced pre-swizzled order.
// h ping-pongs in global memory; one atomic grid barrier per step.
// ---------------------------------------------------------------------------
__global__ __launch_bounds__(256) void lstm_rec(const float* __restrict__ gx,
                                                const U128* __restrict__ wsw,
                                                float* __restrict__ h0buf,
                                                float* __restrict__ h1buf,
                                                float* __restrict__ last,
                                                int* __restrict__ bar) {
  __shared__ float hs[HH];

  const int tid  = threadIdx.x;
  const int wg   = blockIdx.x;
  const int wave = tid >> 5, lane = tid & 31;
  const int t4   = lane & 3;
  const int R    = wave * 8 + (lane >> 2);    // 0..63 row-in-block
  const int gate = R & 3, unitL = R >> 2;     // unitL 0..15
  const int row  = gate * HH + wg * 16 + unitL;
  const int kbase = t4 * 384;
  const bool leader = (lane & 15) == 0;       // one per unit (2 units / wave)
  const int  base   = lane & 16;

  int* cnt           = bar;
  volatile int* gen  = bar + 1;

  float c = 0.0f;                             // carried cell state (leader lanes)

  for (int t = 0; t < BS; ++t) {
    const float* hin  = (t & 1) ? h1buf : h0buf;
    float*       hout = (t & 1) ? h0buf : h1buf;

    // Stage h into LDS (coalesced)
    for (int i = tid; i < HH; i += 256) hs[i] = hin[i];
    __syncthreads();

    // Partial dot: 384 K-elements per lane, 8 bf16 per 16B load
    float acc = 0.0f;
#pragma unroll 4
    for (int it = 0; it < 48; ++it) {
      if (it + 8 < 48)
        __builtin_prefetch(&wsw[((size_t)(it + 8) * RWG + wg) * 256 + tid], 0, 0);
      U128 w = wsw[((size_t)it * RWG + wg) * 256 + tid];
      int k = kbase + it * 8;
      acc = fmaf(bf16lo_to_f32(w.x), hs[k + 0], acc);
      acc = fmaf(bf16hi_to_f32(w.x), hs[k + 1], acc);
      acc = fmaf(bf16lo_to_f32(w.y), hs[k + 2], acc);
      acc = fmaf(bf16hi_to_f32(w.y), hs[k + 3], acc);
      acc = fmaf(bf16lo_to_f32(w.z), hs[k + 4], acc);
      acc = fmaf(bf16hi_to_f32(w.z), hs[k + 5], acc);
      acc = fmaf(bf16lo_to_f32(w.w), hs[k + 6], acc);
      acc = fmaf(bf16hi_to_f32(w.w), hs[k + 7], acc);
    }
    // Reduce the 4 lanes of each row
    acc += __shfl_xor(acc, 1, 32);
    acc += __shfl_xor(acc, 2, 32);

    float g = gx[(size_t)t * G4 + row] + acc;

    // Gather the 4 gates of this half-wave's unit onto the leader lane
    float gi = __shfl(g, base + 0, 32);
    float gf = __shfl(g, base + 4, 32);
    float gg = __shfl(g, base + 8, 32);
    float go = __shfl(g, base + 12, 32);

    if (leader) {
      gi = 1.0f / (1.0f + __expf(-gi));
      gf = 1.0f / (1.0f + __expf(-gf));
      gg = tanhf(gg);
      go = 1.0f / (1.0f + __expf(-go));
      c = gf * c + gi * gg;
      float h = go * tanhf(c);
      int ug = wg * 16 + wave * 2 + (lane >> 4);
      hout[ug] = h;
      if (((t + 1) & (SS - 1)) == 0)
        last[(size_t)(((t + 1) >> 9) - 1) * HH + ug] = h;
    }

    // Grid-wide barrier: all h writes visible before next step's reads
    __threadfence();
    __syncthreads();
    if (tid == 0) {
      int g0 = *gen;
      if (atomicAdd(cnt, 1) == RWG - 1) {
        *cnt = 0;
        __threadfence();
        atomicAdd((int*)gen, 1);
      } else {
        while (*gen == g0) __builtin_amdgcn_s_sleep(1);
      }
    }
    __syncthreads();
  }
}

// ---------------------------------------------------------------------------
// Phase 3: Mish -> 2-unit linear -> log_softmax for 16 batch rows
// ---------------------------------------------------------------------------
__global__ __launch_bounds__(256) void finish_head(const float* __restrict__ last,
                                                   const float* __restrict__ w_lin,
                                                   const float* __restrict__ b_lin,
                                                   float* __restrict__ out) {
  int wave = threadIdx.x >> 5, lane = threadIdx.x & 31;
  for (int b = wave; b < BB; b += 8) {
    float d0 = 0.0f, d1 = 0.0f;
    for (int j = lane; j < HH; j += 32) {
      float x  = last[(size_t)b * HH + j];
      float sp = logf(1.0f + __expf(x));          // softplus
      float a  = x * tanhf(sp);                   // mish
      d0 = fmaf(a, w_lin[j], d0);
      d1 = fmaf(a, w_lin[HH + j], d1);
    }
#pragma unroll
    for (int o = 16; o >= 1; o >>= 1) {
      d0 += __shfl_xor(d0, o, 32);
      d1 += __shfl_xor(d1, o, 32);
    }
    if (lane == 0) {
      float l0 = d0 + b_lin[0], l1 = d1 + b_lin[1];
      float m   = fmaxf(l0, l1);
      float lse = m + logf(__expf(l0 - m) + __expf(l1 - m));
      out[b * 2 + 0] = l0 - lse;
      out[b * 2 + 1] = l1 - lse;
    }
  }
}

// ---------------------------------------------------------------------------
// Launch
// ---------------------------------------------------------------------------
extern "C" void kernel_launch(void* const* d_in, const int* in_sizes, int n_in,
                              void* d_out, int out_size, void* d_ws, size_t ws_size,
                              hipStream_t stream) {
  (void)in_sizes; (void)n_in; (void)out_size; (void)ws_size;

  const float* x     = (const float*)d_in[0];   // (16,512,768)
  const float* w_ih  = (const float*)d_in[1];   // (6144,768)
  const float* w_hh  = (const float*)d_in[2];   // (6144,1536)
  const float* b_ih  = (const float*)d_in[3];   // (6144)
  const float* b_hh  = (const float*)d_in[4];   // (6144)
  const float* w_lin = (const float*)d_in[5];   // (2,1536)
  const float* b_lin = (const float*)d_in[6];   // (2)
  float* out = (float*)d_out;                   // (16,2)

  // Workspace layout (all 256B aligned); total ~242.3 MB
  char* ws = (char*)d_ws;
  constexpr size_t OFF_XB   = 0;                                  // 8192*768 bf16
  constexpr size_t OFF_WIHB = OFF_XB   + (size_t)BS * II * 2;     // 6144*768 bf16
  constexpr size_t OFF_WHH  = OFF_WIHB + (size_t)G4 * II * 2;     // 6144*1536 bf16 (swizzled)
  constexpr size_t OFF_GX   = OFF_WHH  + (size_t)G4 * HH * 2;     // 8192*6144 f32
  constexpr size_t OFF_H    = OFF_GX   + (size_t)BS * G4 * 4;     // 2*1536 f32 ping-pong
  constexpr size_t OFF_LAST = OFF_H    + (size_t)2 * HH * 4;      // 16*1536 f32
  constexpr size_t OFF_BAR  = OFF_LAST + (size_t)BB * HH * 4;     // barrier ints

  unsigned short* xb   = (unsigned short*)(ws + OFF_XB);
  unsigned short* wihb = (unsigned short*)(ws + OFF_WIHB);
  unsigned short* whhb = (unsigned short*)(ws + OFF_WHH);
  float* gx   = (float*)(ws + OFF_GX);
  float* hbuf = (float*)(ws + OFF_H);
  float* lastb= (float*)(ws + OFF_LAST);
  int*   bar  = (int*)  (ws + OFF_BAR);

  // Prologue conversions / swizzles / init
  cvt_bf16<<<(BS * II) / 1024, 256, 0, stream>>>(x, xb, BS * II);
  cvt_bf16<<<(G4 * II) / 1024, 256, 0, stream>>>(w_ih, wihb, G4 * II);
  swizzle_whh<<<(48 * RWG * 256) / 256, 256, 0, stream>>>(w_hh, whhb);
  init_state<<<12, 256, 0, stream>>>(hbuf, bar);

  // Phase 1: bf16 WMMA GEMM (8192x6144x768)
  gemm_gx<<<dim3(G4 / 128, BS / 128), 256, 0, stream>>>(xb, wihb, b_ih, b_hh, gx);

  // Phase 2: persistent sequential recurrence
  lstm_rec<<<RWG, 256, 0, stream>>>(gx, (const U128*)whhb, hbuf, hbuf + HH, lastb, bar);

  // Phase 3: head
  finish_head<<<1, 256, 0, stream>>>(lastb, w_lin, b_lin, out);
}
